// TrainableMITG_71975061946752
// MI455X (gfx1250) — matrix operand order repaired and checked by reading, compiled, not run on hardware
//
#include <hip/hip_runtime.h>

// ---------------------------------------------------------------------------
// MI455X (gfx1250) implementation.
// Bandwidth-bound workload (~170MB traffic @ 23.3TB/s). Heavy GEMMs are done
// with v_wmma_f32_16x16x32_f16 (f16 in, f32 accumulate). The broadcast parts
// of the scorer inputs are factored into per-b bias vectors; P_stable
// reduces analytically to the uniform 1/C (softmax of a constant row), so
// the full-sequence mean / Wstab path is skipped entirely.
// ---------------------------------------------------------------------------

typedef _Float16 v16h __attribute__((ext_vector_type(16)));
typedef _Float16 v8h  __attribute__((ext_vector_type(8)));
typedef float    v8f  __attribute__((ext_vector_type(8)));

constexpr int Bn   = 2048;
constexpr int Ln   = 200;
constexpr int Cn   = 100;
constexpr int DIDn = 64;
constexpr int DTXTn= 128;
constexpr int DCTXn= 128;
constexpr int Kn   = 5;
constexpr int Vn   = 2000;
constexpr int DINn = DIDn + DTXTn;   // 192
constexpr int NG   = 3 * DCTXn;      // 384
constexpr int HIDc = 128;
constexpr int MXI  = Bn * Kn;        // 10240 rows for input-gate GEMM
constexpr int MC   = Bn * Cn;        // 204800 rows for scorer GEMMs

// ---- workspace layout (byte offsets, all 256-aligned) ----
constexpr size_t OFF_XK16  = 0;                              // [MXI,192] f16 gathered window
constexpr size_t SZ_XK16   = (size_t)MXI * DINn * 2;
constexpr size_t OFF_WIH16 = OFF_XK16 + SZ_XK16;             // [384,192] f16
constexpr size_t SZ_WIH16  = (size_t)NG * DINn * 2;
constexpr size_t OFF_WHH16 = OFF_WIH16 + SZ_WIH16;           // [384,128] f16
constexpr size_t SZ_WHH16  = (size_t)NG * DCTXn * 2;
constexpr size_t OFF_WT1M  = OFF_WHH16 + SZ_WHH16;           // Wt1[:,128:256] f16 [128,128]
constexpr size_t SZ_WT1M   = (size_t)128 * 128 * 2;
constexpr size_t OFF_WI1M  = OFF_WT1M + SZ_WT1M;             // Wi1[:,64:128]  f16 [128,64]
constexpr size_t SZ_WI1M   = (size_t)128 * 64 * 2;
constexpr size_t OFF_XI    = OFF_WI1M + SZ_WI1M;             // [MXI,384] f32 input gates (+b_ih)
constexpr size_t SZ_XI     = (size_t)MXI * NG * 4;
constexpr size_t OFF_H     = OFF_XI + SZ_XI;                 // [B,128] f32 GRU hidden
constexpr size_t SZ_H      = (size_t)Bn * HIDc * 4;
constexpr size_t OFF_UT    = OFF_H + SZ_H;                   // [B,128] f32 text per-b bias
constexpr size_t SZ_UT     = (size_t)Bn * HIDc * 4;
constexpr size_t OFF_UI    = OFF_UT + SZ_UT;                 // [B,128] f32 id per-b bias
constexpr size_t SZ_UI     = (size_t)Bn * HIDc * 4;
constexpr size_t OFF_ST    = OFF_UI + SZ_UI;                 // [B,C] text scores
constexpr size_t SZ_ST     = (size_t)Bn * Cn * 4;
constexpr size_t OFF_SI    = OFF_ST + SZ_ST;                 // [B,C] id scores
constexpr size_t SZ_SI     = (size_t)Bn * Cn * 4;
constexpr size_t OFF_AL    = OFF_SI + SZ_SI;                 // [B,3] alpha
// total ~24.8 MB

// ---------------------------------------------------------------------------
// WMMA fragment loaders (CDNA5 16x16x32 f16 layouts, wave32)
//   A 16x32 : lane&15 = M ; lanes<16 -> K {k0..k0+7, k0+16..k0+23},
//             lanes>=16 -> K {k0+8..k0+15, k0+24..k0+31}
//   B 32x16 : lane&15 = N ; same K split as A
//   C/D 16x16 f32: elem i -> M = m0 + i + (lane>=16 ? 8 : 0), N = n0 + (lane&15)
// ---------------------------------------------------------------------------
__device__ __forceinline__ v16h a_frag_f16(const _Float16* A, int ld, int m0, int k0, int lane) {
  const int M   = m0 + (lane & 15);
  const int klo = k0 + ((lane & 16) ? 8 : 0);
  const _Float16* p = A + (size_t)M * ld + klo;
  union { v16h v; v8h h[2]; } u;
  u.h[0] = *(const v8h*)(p);
  u.h[1] = *(const v8h*)(p + 16);
  return u.v;
}

__device__ __forceinline__ v16h a_frag_f32(const float* A, int ld, int m0, int k0, int lane) {
  const int M   = m0 + (lane & 15);
  const int klo = k0 + ((lane & 16) ? 8 : 0);
  const float* p = A + (size_t)M * ld + klo;
  v16h v;
#pragma unroll
  for (int i = 0; i < 8; ++i) {
    v[i]     = (_Float16)p[i];
    v[i + 8] = (_Float16)p[i + 16];
  }
  return v;
}

__device__ __forceinline__ v16h b_frag_f16(const _Float16* W, int ld, int n0, int k0, int lane) {
  const int N  = n0 + (lane & 15);
  const int kb = k0 + ((lane & 16) ? 8 : 0);
  const _Float16* p = W + (size_t)N * ld + kb;
  union { v16h v; v8h h[2]; } u;
  u.h[0] = *(const v8h*)(p);
  u.h[1] = *(const v8h*)(p + 16);
  return u.v;
}

__device__ __forceinline__ v8f wmma32(v16h a, v16h b, v8f c) {
  return __builtin_amdgcn_wmma_f32_16x16x32_f16(false, a, false, b, (short)0, c, false, false);
}

// ---------------------------------------------------------------------------
// Kernel 1: convert weights to f16 (re-used many times; keep resident in L2)
// ---------------------------------------------------------------------------
__global__ void k_prep_weights(const float* __restrict__ Wih, const float* __restrict__ Whh,
                               const float* __restrict__ Wt1, const float* __restrict__ Wi1,
                               _Float16* wih16, _Float16* whh16, _Float16* wt1m, _Float16* wi1m) {
  const int idx = blockIdx.x * blockDim.x + threadIdx.x;
  const int n1 = NG * DINn;          // 73728
  const int n2 = n1 + NG * DCTXn;    // 122880
  const int n3 = n2 + 128 * 128;     // 139264
  const int n4 = n3 + 128 * 64;      // 147456
  if (idx < n1) {
    wih16[idx] = (_Float16)Wih[idx];
  } else if (idx < n2) {
    const int j = idx - n1;
    whh16[j] = (_Float16)Whh[j];
  } else if (idx < n3) {
    const int j = idx - n2;                     // Wt1 cand-text block: cols 128..255
    wt1m[j] = (_Float16)Wt1[(j >> 7) * 384 + 128 + (j & 127)];
  } else if (idx < n4) {
    const int j = idx - n3;                     // Wi1 cand-id block: cols 64..127
    wi1m[j] = (_Float16)Wi1[(j >> 6) * 256 + 64 + (j & 63)];
  }
}

// ---------------------------------------------------------------------------
// Kernel 2: gather K-window of seq embeddings into f16 A-matrix [B*K, 192]
// ---------------------------------------------------------------------------
__global__ void k_gather_xk(const float* __restrict__ id_emb, const float* __restrict__ text_emb,
                            const int* __restrict__ lengths, _Float16* __restrict__ xk16) {
  const int idx = blockIdx.x * blockDim.x + threadIdx.x;
  if (idx >= MXI * DINn) return;
  const int d   = idx % DINn;
  const int row = idx / DINn;
  const int b = row / Kn, t = row % Kn;
  const int l = lengths[b] - Kn + t;
  float v;
  if (d < DIDn) v = id_emb[((size_t)b * Ln + l) * DIDn + d];
  else          v = text_emb[((size_t)b * Ln + l) * DTXTn + (d - DIDn)];
  xk16[idx] = (_Float16)v;
}

// ---------------------------------------------------------------------------
// Kernel 3: input-gate GEMM  XI[MXI,384] = xk16 @ W_ih^T + b_ih
// one wave per 16x16 tile, k-loop of 6 WMMAs
// ---------------------------------------------------------------------------
__global__ void __launch_bounds__(256) k_gemm_xi(const _Float16* __restrict__ A,
                                                 const _Float16* __restrict__ W,
                                                 const float* __restrict__ bih,
                                                 float* __restrict__ XI) {
  const int lane = threadIdx.x & 31;
  const int tile = blockIdx.x * 8 + (threadIdx.x >> 5);
  const int mt = tile / (NG / 16), nt = tile % (NG / 16);
  const int m0 = mt * 16, n0 = nt * 16;
  v8f acc = {};
#pragma unroll
  for (int kk = 0; kk < DINn / 32; ++kk) {
    const int k0 = kk * 32;
    v16h a  = a_frag_f16(A, DINn, m0, k0, lane);
    v16h bf = b_frag_f16(W, DINn, n0, k0, lane);
    acc = wmma32(a, bf, acc);
  }
  const int N = n0 + (lane & 15);
  const float bb = bih[N];
#pragma unroll
  for (int i = 0; i < 8; ++i) {
    const int M = m0 + i + ((lane & 16) ? 8 : 0);
    XI[(size_t)M * NG + N] = acc[i] + bb;
  }
}

// ---------------------------------------------------------------------------
// Kernel 4: GRU recurrence. Rows are independent -> each block owns 16 batch
// rows, keeps h in LDS, 5 sequential steps of [16,128]@W_hh^T via WMMA.
// ---------------------------------------------------------------------------
__global__ void __launch_bounds__(256) k_gru(const float* __restrict__ XI,
                                             const _Float16* __restrict__ whh16,
                                             const float* __restrict__ b_hh,
                                             float* __restrict__ H) {
  __shared__ __align__(16) _Float16 hf16[16 * 128];   // h as f16 A-matrix
  __shared__ float hg[16 * 384];                       // h @ W_hh^T + b_hh
  __shared__ float hprev[16 * 128];                    // h as f32
  const int tid  = threadIdx.x;
  const int wave = tid >> 5, lane = tid & 31;
  const int gb0  = blockIdx.x * 16;

  for (int idx = tid; idx < 16 * 128; idx += 256) { hprev[idx] = 0.f; hf16[idx] = (_Float16)0.f; }
  __syncthreads();

  for (int t = 0; t < Kn; ++t) {
    // --- Hg = h @ W_hh^T : 24 N-tiles, 8 waves x 3 tiles, k-loop = 4 ---
    v8f acc[3];
#pragma unroll
    for (int q = 0; q < 3; ++q) { v8f z = {}; acc[q] = z; }
#pragma unroll
    for (int kk = 0; kk < 4; ++kk) {
      const int k0 = kk * 32;
      v16h a = a_frag_f16(hf16, 128, 0, k0, lane);
#pragma unroll
      for (int q = 0; q < 3; ++q) {
        const int n0 = (wave * 3 + q) * 16;
        v16h bf = b_frag_f16(whh16, 128, n0, k0, lane);
        acc[q] = wmma32(a, bf, acc[q]);
      }
    }
#pragma unroll
    for (int q = 0; q < 3; ++q) {
      const int N = (wave * 3 + q) * 16 + (lane & 15);
      const float bh = b_hh[N];
#pragma unroll
      for (int i = 0; i < 8; ++i) {
        const int Mr = i + ((lane & 16) ? 8 : 0);
        hg[Mr * 384 + N] = acc[q][i] + bh;
      }
    }
    __syncthreads();
    // --- gates ---
    for (int idx = tid; idx < 16 * 128; idx += 256) {
      const int r = idx >> 7, j = idx & 127;
      const size_t base = ((size_t)(gb0 + r) * Kn + t) * NG;
      const float ir = XI[base + j], iz = XI[base + 128 + j], inn = XI[base + 256 + j];
      const float hr = hg[r * 384 + j], hz = hg[r * 384 + 128 + j], hn = hg[r * 384 + 256 + j];
      const float rg = 1.f / (1.f + expf(-(ir + hr)));
      const float zg = 1.f / (1.f + expf(-(iz + hz)));
      const float ng = tanhf(inn + rg * hn);
      const float hv = (1.f - zg) * ng + zg * hprev[idx];
      hprev[idx] = hv;
      hf16[idx]  = (_Float16)hv;
    }
    __syncthreads();
  }
  for (int idx = tid; idx < 16 * 128; idx += 256)
    H[(size_t)gb0 * HIDc + idx] = hprev[idx];
}

// ---------------------------------------------------------------------------
// Kernel 5: per-b bias vectors u_t, u_i (broadcast parts of the scorer MLPs)
// and mixture weights alpha. Small matvecs -> plain VALU.
// ---------------------------------------------------------------------------
__global__ void __launch_bounds__(128) k_pervec(const float* __restrict__ H, const int* __restrict__ lengths,
                                                const float* __restrict__ id_emb, const float* __restrict__ text_emb,
                                                const float* __restrict__ Wt1, const float* __restrict__ bt1,
                                                const float* __restrict__ Wi1, const float* __restrict__ bi1,
                                                const float* __restrict__ Wa1, const float* __restrict__ ba1,
                                                const float* __restrict__ Wa2, const float* __restrict__ ba2,
                                                float* __restrict__ UT, float* __restrict__ UI,
                                                float* __restrict__ ALPHA) {
  __shared__ float sh_h[128], sh_lt[128], sh_lid[64], sh_a1[64], sh_lg[3];
  const int b = blockIdx.x, j = threadIdx.x;
  const int li = lengths[b] - 1;
  sh_h[j]  = H[(size_t)b * HIDc + j];
  sh_lt[j] = text_emb[((size_t)b * Ln + li) * DTXTn + j];
  if (j < 64) sh_lid[j] = id_emb[((size_t)b * Ln + li) * DIDn + j];
  __syncthreads();
  {
    const float* wr = Wt1 + (size_t)j * 384;
    float acc = bt1[j];
    for (int k = 0; k < 128; ++k) acc += wr[k] * sh_lt[k];
    for (int k = 0; k < 128; ++k) acc += wr[256 + k] * sh_h[k];
    UT[(size_t)b * HIDc + j] = acc;
  }
  {
    const float* wr = Wi1 + (size_t)j * 256;
    float acc = bi1[j];
    for (int k = 0; k < 64;  ++k) acc += wr[k] * sh_lid[k];
    for (int k = 0; k < 128; ++k) acc += wr[128 + k] * sh_h[k];
    UI[(size_t)b * HIDc + j] = acc;
  }
  if (j < 64) {
    const float* wr = Wa1 + (size_t)j * 128;
    float acc = ba1[j];
    for (int k = 0; k < 128; ++k) acc += wr[k] * sh_h[k];
    sh_a1[j] = acc > 0.f ? acc : 0.f;
  }
  __syncthreads();
  if (j < 3) {
    const float* wr = Wa2 + (size_t)j * 64;
    float acc = ba2[j];
    for (int k = 0; k < 64; ++k) acc += wr[k] * sh_a1[k];
    sh_lg[j] = acc;
  }
  __syncthreads();
  if (j == 0) {
    const float m = fmaxf(sh_lg[0], fmaxf(sh_lg[1], sh_lg[2]));
    const float e0 = expf(sh_lg[0] - m), e1 = expf(sh_lg[1] - m), e2 = expf(sh_lg[2] - m);
    const float s = e0 + e1 + e2;
    ALPHA[b * 3 + 0] = e0 / s; ALPHA[b * 3 + 1] = e1 / s; ALPHA[b * 3 + 2] = e2 / s;
  }
}

// ---------------------------------------------------------------------------
// Kernel 6: fused scorer GEMM. One wave = 16 rows x all 128 hidden units:
// 8 N-tiles, KT/32 k-steps; epilogue adds per-b bias, ReLU, dot W2 with a
// 16-lane shfl_xor reduction; writes score per (b,c) row.
// ---------------------------------------------------------------------------
template<int KT>
__global__ void __launch_bounds__(256) k_score(const float* __restrict__ Aemb,
                                               const _Float16* __restrict__ Wm,
                                               const float* __restrict__ U,
                                               const float* __restrict__ w2,
                                               const float* __restrict__ b2,
                                               float* __restrict__ S) {
  const int lane = threadIdx.x & 31;
  const int m0   = (blockIdx.x * 8 + (threadIdx.x >> 5)) * 16;
  v8f acc[8];
#pragma unroll
  for (int q = 0; q < 8; ++q) { v8f z = {}; acc[q] = z; }
#pragma unroll
  for (int kk = 0; kk < KT / 32; ++kk) {
    const int k0 = kk * 32;
    v16h a = a_frag_f32(Aemb, KT, m0, k0, lane);   // f32 -> f16 on the fly
#pragma unroll
    for (int q = 0; q < 8; ++q) {
      v16h bf = b_frag_f16(Wm, KT, q * 16, k0, lane);
      acc[q] = wmma32(a, bf, acc[q]);
    }
  }
  const int nl = lane & 15;
  float w2v[8];
#pragma unroll
  for (int q = 0; q < 8; ++q) w2v[q] = w2[q * 16 + nl];
  const float b2v = b2[0];
  float partial[8];
#pragma unroll
  for (int i = 0; i < 8; ++i) {
    const int M  = m0 + i + ((lane & 16) ? 8 : 0);
    const int bb = M / Cn;
    const float* Ub = U + (size_t)bb * HIDc + nl;
    float p = 0.f;
#pragma unroll
    for (int q = 0; q < 8; ++q) {
      float v = acc[q][i] + Ub[q * 16];
      p += (v > 0.f ? v : 0.f) * w2v[q];
    }
#pragma unroll
    for (int msk = 8; msk >= 1; msk >>= 1)
      p += __shfl_xor(p, msk, 16);
    partial[i] = p;
  }
  if (nl == 0) {
#pragma unroll
    for (int i = 0; i < 8; ++i) {
      const int M = m0 + i + ((lane & 16) ? 8 : 0);
      S[M] = partial[i] + b2v;
    }
  }
}

// ---------------------------------------------------------------------------
// Kernel 7: co-occurrence gather, three softmaxes over C, mixture + uniform
// P_stable (=1/C analytically), clip, renormalize.
// ---------------------------------------------------------------------------
__device__ __forceinline__ float blk_max128(float v, float* red) {
  const int t = threadIdx.x;
  red[t] = v; __syncthreads();
#pragma unroll
  for (int s = 64; s > 0; s >>= 1) { if (t < s) red[t] = fmaxf(red[t], red[t + s]); __syncthreads(); }
  const float r = red[0]; __syncthreads();
  return r;
}
__device__ __forceinline__ float blk_sum128(float v, float* red) {
  const int t = threadIdx.x;
  red[t] = v; __syncthreads();
#pragma unroll
  for (int s = 64; s > 0; s >>= 1) { if (t < s) red[t] += red[t + s]; __syncthreads(); }
  const float r = red[0]; __syncthreads();
  return r;
}

__global__ void __launch_bounds__(128) k_final(const int* __restrict__ seq_items, const int* __restrict__ lengths,
                                               const int* __restrict__ cand_ids, const float* __restrict__ co_table,
                                               const float* __restrict__ ST, const float* __restrict__ SI,
                                               const float* __restrict__ ALPHA,
                                               const float* __restrict__ beta_p, const float* __restrict__ lTc,
                                               const float* __restrict__ lTt, const float* __restrict__ lTi,
                                               float* __restrict__ out) {
  __shared__ float red[128];
  const int b = blockIdx.x, t = threadIdx.x;
  const bool valid = t < Cn;
  const float beta = beta_p[0];
  const float iTc = 1.f / expf(lTc[0]);
  const float iTt = 1.f / expf(lTt[0]);
  const float iTi = 1.f / expf(lTi[0]);
  const int li   = lengths[b] - 1;
  const int item = seq_items[(size_t)b * Ln + li];
  float xc = -3.0e38f, xt = -3.0e38f, xi = -3.0e38f;
  if (valid) {
    const int cid = cand_ids[b * Cn + t];
    xc = beta * co_table[(size_t)item * Vn + cid] * iTc;
    xt = ST[b * Cn + t] * iTt;
    xi = SI[b * Cn + t] * iTi;
  }
  const float mc = blk_max128(xc, red);
  const float mt = blk_max128(xt, red);
  const float mi = blk_max128(xi, red);
  const float ec = valid ? expf(xc - mc) : 0.f;
  const float et = valid ? expf(xt - mt) : 0.f;
  const float ei = valid ? expf(xi - mi) : 0.f;
  const float sc = blk_sum128(ec, red);
  const float st = blk_sum128(et, red);
  const float si = blk_sum128(ei, red);
  const float a0 = ALPHA[b * 3 + 0], a1 = ALPHA[b * 3 + 1], a2 = ALPHA[b * 3 + 2];
  float pm = a0 * (ec / sc) + a1 * (et / st) + a2 * (ei / si) + (1.0f / Cn);
  pm = fmaxf(pm, 1e-12f);
  const float s = blk_sum128(valid ? pm : 0.f, red);
  if (valid) out[(size_t)b * Cn + t] = pm / s;
}

// ---------------------------------------------------------------------------
extern "C" void kernel_launch(void* const* d_in, const int* in_sizes, int n_in,
                              void* d_out, int out_size, void* d_ws, size_t ws_size,
                              hipStream_t stream) {
  (void)in_sizes; (void)n_in; (void)out_size; (void)ws_size;
  const int*   seq_items     = (const int*)  d_in[0];
  const float* seq_id_emb    = (const float*)d_in[1];
  const float* seq_text_emb  = (const float*)d_in[2];
  const int*   lengths       = (const int*)  d_in[3];
  const int*   cand_ids      = (const int*)  d_in[4];
  const float* cand_id_emb   = (const float*)d_in[5];
  const float* cand_text_emb = (const float*)d_in[6];
  const float* co_table      = (const float*)d_in[7];
  const float* W_ih          = (const float*)d_in[8];
  const float* W_hh          = (const float*)d_in[9];
  const float* b_ih          = (const float*)d_in[10];
  const float* b_hh          = (const float*)d_in[11];
  const float* Wt1           = (const float*)d_in[12];
  const float* bt1           = (const float*)d_in[13];
  const float* Wt2           = (const float*)d_in[14];
  const float* bt2           = (const float*)d_in[15];
  const float* Wi1           = (const float*)d_in[16];
  const float* bi1           = (const float*)d_in[17];
  const float* Wi2           = (const float*)d_in[18];
  const float* bi2           = (const float*)d_in[19];
  const float* Wa1           = (const float*)d_in[20];
  const float* ba1           = (const float*)d_in[21];
  const float* Wa2           = (const float*)d_in[22];
  const float* ba2           = (const float*)d_in[23];
  // d_in[24] = Wstab: unused (P_stable is analytically uniform 1/C)
  const float* beta_c        = (const float*)d_in[25];
  const float* logT_c        = (const float*)d_in[26];
  const float* logT_t        = (const float*)d_in[27];
  const float* logT_id       = (const float*)d_in[28];

  char* ws = (char*)d_ws;
  _Float16* XK16  = (_Float16*)(ws + OFF_XK16);
  _Float16* WIH16 = (_Float16*)(ws + OFF_WIH16);
  _Float16* WHH16 = (_Float16*)(ws + OFF_WHH16);
  _Float16* WT1M  = (_Float16*)(ws + OFF_WT1M);
  _Float16* WI1M  = (_Float16*)(ws + OFF_WI1M);
  float* XI    = (float*)(ws + OFF_XI);
  float* Hbuf  = (float*)(ws + OFF_H);
  float* UT    = (float*)(ws + OFF_UT);
  float* UI    = (float*)(ws + OFF_UI);
  float* STb   = (float*)(ws + OFF_ST);
  float* SIb   = (float*)(ws + OFF_SI);
  float* ALPHA = (float*)(ws + OFF_AL);
  float* out   = (float*)d_out;

  k_prep_weights<<<576, 256, 0, stream>>>(W_ih, W_hh, Wt1, Wi1, WIH16, WHH16, WT1M, WI1M);
  k_gather_xk<<<(MXI * DINn + 255) / 256, 256, 0, stream>>>(seq_id_emb, seq_text_emb, lengths, XK16);
  k_gemm_xi<<<(MXI / 16) * (NG / 16) / 8, 256, 0, stream>>>(XK16, WIH16, b_ih, XI);
  k_gru<<<Bn / 16, 256, 0, stream>>>(XI, WHH16, b_hh, Hbuf);
  k_pervec<<<Bn, 128, 0, stream>>>(Hbuf, lengths, seq_id_emb, seq_text_emb,
                                   Wt1, bt1, Wi1, bi1, Wa1, ba1, Wa2, ba2, UT, UI, ALPHA);
  k_score<128><<<(MC / 16) / 8, 256, 0, stream>>>(cand_text_emb, WT1M, UT, Wt2, bt2, STb);
  k_score<64 ><<<(MC / 16) / 8, 256, 0, stream>>>(cand_id_emb,  WI1M, UI, Wi2, bi2, SIb);
  k_final<<<Bn, 128, 0, stream>>>(seq_items, lengths, cand_ids, co_table, STb, SIb, ALPHA,
                                  beta_c, logT_c, logT_t, logT_id, out);
}
